// Transducer_23673859736084
// MI455X (gfx1250) — matrix operand order repaired and checked by reading, not compiled, not run on hardware
//
#include <hip/hip_runtime.h>
#include <hip/hip_bf16.h>

// Problem constants (match reference)
#define B_   8
#define T_   300
#define U_   80
#define UP1  81
#define V_   256
#define E_   512
#define IN_  512

typedef __attribute__((ext_vector_type(16))) _Float16 v16h;
typedef __attribute__((ext_vector_type(8)))  _Float16 v8h;
typedef __attribute__((ext_vector_type(8)))  float    v8f;

__device__ __forceinline__ float wave_red_max(float v) {
#pragma unroll
  for (int m = 16; m >= 1; m >>= 1) v = fmaxf(v, __shfl_xor(v, m, 32));
  return v;
}
__device__ __forceinline__ float wave_red_sum(float v) {
#pragma unroll
  for (int m = 16; m >= 1; m >>= 1) v += __shfl_xor(v, m, 32);
  return v;
}

// ---------------------------------------------------------------------------
// 1) Projection GEMMs (fp32): out[m,i] = dot(X[m,:512], W1[i, wofs:wofs+512])
//    W1 is (512, 1024) row-major.  One block per output row m.
// ---------------------------------------------------------------------------
__global__ void __launch_bounds__(256)
proj_gemm(const float* __restrict__ X, const float* __restrict__ W1,
          float* __restrict__ out, int wofs) {
  __shared__ float xs[512];
  const int m   = blockIdx.x;
  const int tid = threadIdx.x;
  xs[tid]       = X[(size_t)m * 512 + tid];
  xs[tid + 256] = X[(size_t)m * 512 + tid + 256];
  __syncthreads();
  for (int i = tid; i < 512; i += 256) {
    const float4* w = (const float4*)(W1 + (size_t)i * 1024 + wofs);
    float acc = 0.f;
#pragma unroll 8
    for (int j = 0; j < 128; ++j) {
      float4 wv = w[j];
      acc += xs[4 * j + 0] * wv.x + xs[4 * j + 1] * wv.y +
             xs[4 * j + 2] * wv.z + xs[4 * j + 3] * wv.w;
    }
    out[(size_t)m * 512 + i] = acc;
  }
}

// ---------------------------------------------------------------------------
// 2) W2 fp32 -> f16 (L2-resident operand for all joint workgroups)
// ---------------------------------------------------------------------------
__global__ void cvt_w2(const float* __restrict__ W2, _Float16* __restrict__ W2h, int n) {
  int i = blockIdx.x * blockDim.x + threadIdx.x;
  if (i < n) W2h[i] = (_Float16)W2[i];
}

// ---------------------------------------------------------------------------
// 3) Fused joint + log-softmax kernel.  1 WG per (b,t), 256 threads (8 waves).
//    Per 16-row u-chunk: A = tanh(h_enc + h_dec + b1) -> LDS (f16),
//    logits(16x256) via v_wmma_f32_16x16x32_f16, then per-row logsumexp and
//    emit only blank_lp / label_lp.
// ---------------------------------------------------------------------------
__global__ void __launch_bounds__(256)
joint_kernel(const float* __restrict__ henc_all, const float* __restrict__ hdec_all,
             const float* __restrict__ b1, const _Float16* __restrict__ W2h,
             const float* __restrict__ b2, const int* __restrict__ targets,
             float* __restrict__ blankp, float* __restrict__ labelp) {
  __shared__ _Float16 Ash[16 * 520];   // 16 rows, stride 520 halves (16B-aligned rows)
  __shared__ float    Lgs[16 * 260];   // 16 rows x 256 logits, stride 260 f32

  const int bt   = blockIdx.x;
  const int b    = bt / T_;
  const int tid  = threadIdx.x;
  const int lane = tid & 31;
  const int w    = tid >> 5;           // wave id 0..7
  const int mlane = lane & 15;
  const int hb    = lane >> 4;         // half-wave: 0 or 1

  const float* henc = henc_all + (size_t)bt * IN_;

  for (int chunk = 0; chunk < 6; ++chunk) {
    // ---- build f16 A tile: rows u = chunk*16 .. chunk*16+15 (pad with 0) ----
    for (int idx = tid; idx < 16 * IN_; idx += 256) {
      int r = idx >> 9;
      int k = idx & (IN_ - 1);
      int u = chunk * 16 + r;
      float v = 0.f;
      if (u < UP1)
        v = tanhf(henc[k] + hdec_all[(size_t)(b * UP1 + u) * IN_ + k] + b1[k]);
      Ash[r * 520 + k] = (_Float16)v;
    }
    __syncthreads();

    // ---- WMMA: each wave owns n-tiles {2w, 2w+1}, K = 512 in 16 steps ----
#pragma unroll
    for (int q = 0; q < 2; ++q) {
      const int nt = 2 * w + q;
      const int n  = nt * 16 + mlane;  // global output column (0..255)
      v8f acc = {};
      // A layout (16-bit 16x32): lane m, element i = K (hb*16 + i), matching B.
      const _Float16* arow = Ash + mlane * 520 + hb * 16;
      // B layout (32x16 f16): lane n = L%16, element i = K (hb*16 + i) -> 16
      // contiguous halves of W2 row n (B[k,n] = W2[n,k]).
      const _Float16* brow = W2h + (size_t)n * IN_ + hb * 16;
#pragma unroll
      for (int ks = 0; ks < 16; ++ks) {
        v8h a0 = *(const v8h*)(arow + ks * 32);
        v8h a1 = *(const v8h*)(arow + ks * 32 + 8);
        v16h afrag = __builtin_shufflevector(a0, a1, 0, 1, 2, 3, 4, 5, 6, 7,
                                             8, 9, 10, 11, 12, 13, 14, 15);
        v16h bfrag = *(const v16h*)(brow + ks * 32);
        acc = __builtin_amdgcn_wmma_f32_16x16x32_f16(
            /*neg_a=*/false, afrag, /*neg_b=*/false, bfrag,
            /*c_mod=*/(short)0, acc, /*reuse_a=*/false, /*reuse_b=*/false);
      }
      const float bias = b2[n];
#pragma unroll
      for (int r = 0; r < 8; ++r) {
        int mrow = r + 8 * hb;         // C/D layout: VGPR r -> row r (+8 for lanes 16-31)
        Lgs[mrow * 260 + n] = acc[r] + bias;
      }
    }
    __syncthreads();

    // ---- per-row logsumexp over V=256; emit blank_lp / label_lp ----
#pragma unroll
    for (int q = 0; q < 2; ++q) {
      const int rr = 2 * w + q;        // row within chunk handled by this wave
      const int u  = chunk * 16 + rr;
      if (u < UP1) {
        float x[8];
        float mx = -3.4e38f;
#pragma unroll
        for (int j = 0; j < 8; ++j) {
          x[j] = Lgs[rr * 260 + lane + 32 * j];
          mx = fmaxf(mx, x[j]);
        }
        mx = wave_red_max(mx);
        float s = 0.f;
#pragma unroll
        for (int j = 0; j < 8; ++j) s += expf(x[j] - mx);
        s = wave_red_sum(s);
        float lse = mx + logf(s);
        if (lane == 0) {
          blankp[(size_t)bt * UP1 + u] = Lgs[rr * 260 + 0] - lse;
          if (u < U_) {
            int tgt = targets[b * U_ + u];
            labelp[(size_t)bt * U_ + u] = Lgs[rr * 260 + tgt] - lse;
          }
        }
      }
    }
    __syncthreads();
  }
}

// ---------------------------------------------------------------------------
// 4) Alpha DP, anti-diagonal wavefront.  1 WG per batch, thread = u lane.
//    alpha[t][u] = logaddexp(alpha[t-1][u] + blank[t-1,u],
//                            alpha[t][u-1] + label[t,u-1])
// ---------------------------------------------------------------------------
__global__ void __launch_bounds__(128)
dp_kernel(const float* __restrict__ blankp, const float* __restrict__ labelp,
          const int* __restrict__ inputs_length, const int* __restrict__ targets_length,
          float* __restrict__ llv) {
  __shared__ float Acur[UP1];
  __shared__ float llsh;
  const int b  = blockIdx.x;
  const int u  = threadIdx.x;
  const int tl = inputs_length[b] - 1;
  const int ul = targets_length[b];
  const float* blk = blankp + (size_t)b * T_ * UP1;
  const float* lab = labelp + (size_t)b * T_ * U_;

  for (int d = 0; d <= (T_ - 1) + U_; ++d) {
    const int t = d - u;
    const bool valid = (u <= U_) && (t >= 0) && (t < T_);
    float nv = 0.f;
    if (valid) {
      if (t == 0 && u == 0) {
        nv = 0.f;
      } else if (u == 0) {
        nv = Acur[0] + blk[(t - 1) * UP1];
      } else if (t == 0) {
        nv = Acur[u - 1] + lab[u - 1];
      } else {
        float a = Acur[u]     + blk[(t - 1) * UP1 + u];
        float c = Acur[u - 1] + lab[t * U_ + (u - 1)];
        float m = fmaxf(a, c);
        nv = m + log1pf(expf(-fabsf(a - c)));
      }
    }
    __syncthreads();                 // all reads of Acur done
    if (valid) {
      Acur[u] = nv;
      if (t == tl && u == ul) llsh = nv + blk[t * UP1 + u];
    }
    __syncthreads();                 // writes visible for next diagonal
  }
  if (u == 0) llv[b] = llsh;
}

// ---------------------------------------------------------------------------
// 5) loss = -mean(ll)
// ---------------------------------------------------------------------------
__global__ void finalize_kernel(const float* __restrict__ llv, float* __restrict__ out) {
  float v = (threadIdx.x < B_) ? llv[threadIdx.x] : 0.f;
  v = wave_red_sum(v);
  if (threadIdx.x == 0) out[0] = -v / (float)B_;
}

// ---------------------------------------------------------------------------
extern "C" void kernel_launch(void* const* d_in, const int* in_sizes, int n_in,
                              void* d_out, int out_size, void* d_ws, size_t ws_size,
                              hipStream_t stream) {
  (void)in_sizes; (void)n_in; (void)out_size; (void)ws_size;
  const float* enc     = (const float*)d_in[0];  // (8,300,512)
  const float* dec     = (const float*)d_in[1];  // (8,81,512)
  const float* W1      = (const float*)d_in[2];  // (512,1024)
  const float* b1      = (const float*)d_in[3];  // (512)
  const float* W2      = (const float*)d_in[4];  // (256,512)
  const float* b2      = (const float*)d_in[5];  // (256)
  const int*   targets = (const int*)d_in[6];    // (8,80)
  const int*   ilen    = (const int*)d_in[7];    // (8)
  const int*   tlen    = (const int*)d_in[8];    // (8)

  float* ws     = (float*)d_ws;
  float* henc   = ws;                                   // 2400*512
  float* hdec   = henc   + (size_t)B_ * T_  * IN_;      // 648*512
  float* blankp = hdec   + (size_t)B_ * UP1 * IN_;      // 8*300*81
  float* labelp = blankp + (size_t)B_ * T_  * UP1;      // 8*300*80
  float* llv    = labelp + (size_t)B_ * T_  * U_;       // 8
  _Float16* W2h = (_Float16*)(llv + 8);                 // 256*512 halves (32B aligned)

  proj_gemm<<<B_ * T_,  256, 0, stream>>>(enc, W1, henc, 0);
  proj_gemm<<<B_ * UP1, 256, 0, stream>>>(dec, W1, hdec, E_);
  cvt_w2<<<(V_ * IN_ + 255) / 256, 256, 0, stream>>>(W2, W2h, V_ * IN_);
  joint_kernel<<<B_ * T_, 256, 0, stream>>>(henc, hdec, b1, W2h, b2, targets,
                                            blankp, labelp);
  dp_kernel<<<B_, 128, 0, stream>>>(blankp, labelp, ilen, tlen, llv);
  finalize_kernel<<<1, 32, 0, stream>>>(llv, (float*)d_out);
}
